// InterModalityUpdate_43404939493743
// MI455X (gfx1250) — matrix operand order repaired and checked by reading, not compiled, not run on hardware
//
#include <hip/hip_runtime.h>

// ---------------------------------------------------------------------------
// InterModalityUpdate for MI455X (gfx1250, wave32, WMMA bf16).
// fp32->bf16 convert (+relu) -> 5 sliced projection GEMMs (masked, bias) ->
// 2 two-pass attentions (LDS-resident S^T, ballot masking) -> 2 output GEMMs.
// GEMM double-buffers LDS tiles with global_load_async_to_lds_b128 and
// partial s_wait_asynccnt waits (ASYNCcnt completes in order) so HBM traffic
// for tile i+1 overlaps the 8 WMMAs of tile i.
// ---------------------------------------------------------------------------

typedef __bf16 bf16;
typedef __attribute__((ext_vector_type(16))) __bf16        v16bf;
typedef __attribute__((ext_vector_type(8)))  float         v8f;
typedef __attribute__((ext_vector_type(4)))  unsigned int  u32x4;

#define NO_   512
#define O_    1024
#define M_    4096           // B*NO
#define DH_   64
#define SCALE_ 0.125f        // 1/sqrt(64)
#define NEGV  (-1e9f)

union FragBF { v16bf v; unsigned int u[8]; };

static __device__ __forceinline__ unsigned short f2bfu(float f) {
  bf16 b = (bf16)f;
  unsigned short u;
  __builtin_memcpy(&u, &b, 2);
  return u;
}

// Async global->LDS copy (CDNA5 VGLOBAL async op, tracked by ASYNCcnt).
// Generic LDS pointers carry the LDS byte address in the low 32 bits.
static __device__ __forceinline__ void async_copy_b128(void* lds, const void* g) {
  unsigned lds_off = (unsigned)(unsigned long long)(size_t)lds;
  unsigned long long ga = (unsigned long long)(size_t)g;
  asm volatile("global_load_async_to_lds_b128 %0, %1, off"
               :: "v"(lds_off), "v"(ga) : "memory");
}
static __device__ __forceinline__ void wait_async0() {
  asm volatile("s_wait_asynccnt 0x0" ::: "memory");
}
// wait until at most the 4 most recently issued async ops remain outstanding
static __device__ __forceinline__ void wait_async4() {
  asm volatile("s_wait_asynccnt 0x4" ::: "memory");
}

// ------------------------- fp32 -> bf16 converters -------------------------

__global__ void cvt_bf16_kernel(const float* __restrict__ src,
                                bf16* __restrict__ dst, int n, int do_relu) {
  int idx = (blockIdx.x * blockDim.x + threadIdx.x) * 4;
  if (idx >= n) return;
  float4 f = *(const float4*)(src + idx);
  if (do_relu) {
    f.x = fmaxf(f.x, 0.0f); f.y = fmaxf(f.y, 0.0f);
    f.z = fmaxf(f.z, 0.0f); f.w = fmaxf(f.w, 0.0f);
  }
  uint2 r;
  r.x = (unsigned)f2bfu(f.x) | ((unsigned)f2bfu(f.y) << 16);
  r.y = (unsigned)f2bfu(f.z) | ((unsigned)f2bfu(f.w) << 16);
  *(uint2*)(dst + idx) = r;
}

// copy fp32 v [4096,1024] into bf16 cat[:,0:1024] (row stride 3072)
__global__ void cvt_v_cat_kernel(const float* __restrict__ v,
                                 bf16* __restrict__ cat) {
  int idx = (blockIdx.x * blockDim.x + threadIdx.x) * 4;
  int row = idx >> 10;
  int col = idx & 1023;
  float4 f = *(const float4*)(v + idx);
  uint2 r;
  r.x = (unsigned)f2bfu(f.x) | ((unsigned)f2bfu(f.y) << 16);
  r.y = (unsigned)f2bfu(f.z) | ((unsigned)f2bfu(f.w) << 16);
  *(uint2*)(cat + (size_t)row * 3072 + col) = r;
}

// ------------------------------ tiled GEMM ---------------------------------
// C[M=4096, N=1024] = A[M,K] * B[K,N] (+bias[n]) (*rowmask[m]); bf16 in,
// bf16 or f32 out. Block tile 128x128, BK=32, 8 waves of 32x64.
// Double-buffered async global->LDS staging.

template <bool MASKED, bool OUTBF16>
__global__ __launch_bounds__(256) void gemm_bf16_kernel(
    const bf16* __restrict__ A, int lda,
    const bf16* __restrict__ Bw, int ldb,
    const float* __restrict__ bias,
    const int* __restrict__ rowmask,
    void* __restrict__ Out, int ldo, int K) {
  __shared__ __align__(16) unsigned short As[2][128 * 40];
  __shared__ __align__(16) unsigned short Bs[2][32 * 136];
  const int tid = threadIdx.x;
  const int lane = tid & 31, wave = tid >> 5;
  const int wm = wave >> 1, wn = wave & 1;   // 4 x 2 wave grid
  const int mBase = blockIdx.y * 128, nBase = blockIdx.x * 128;
  const int lq = lane & 15, half = lane >> 4;

  const v8f zero = {0.f, 0.f, 0.f, 0.f, 0.f, 0.f, 0.f, 0.f};
  v8f c[2][4];
#pragma unroll
  for (int i = 0; i < 2; ++i)
#pragma unroll
    for (int j = 0; j < 4; ++j) c[i][j] = zero;

  const int ra = tid >> 1, sa0 = (tid & 1) * 2;    // A: 2 b128 chunks/thread
  const int rb = tid >> 4, sb0 = (tid & 15);       // B: 2 b128 chunks/thread

  // issue the 4 async b128 copies staging tile k0 into buffer `buf`
  auto issue = [&](int k0, int buf) {
    async_copy_b128(&As[buf][ra * 40 + sa0 * 8],
                    A + (size_t)(mBase + ra) * lda + k0 + sa0 * 8);
    async_copy_b128(&As[buf][ra * 40 + (sa0 + 1) * 8],
                    A + (size_t)(mBase + ra) * lda + k0 + (sa0 + 1) * 8);
    async_copy_b128(&Bs[buf][rb * 136 + sb0 * 8],
                    Bw + (size_t)(k0 + rb) * ldb + nBase + sb0 * 8);
    async_copy_b128(&Bs[buf][(rb + 16) * 136 + sb0 * 8],
                    Bw + (size_t)(k0 + rb + 16) * ldb + nBase + sb0 * 8);
  };
  // 8 WMMAs on buffer `buf`
  auto compute = [&](int buf) {
    FragBF af[2], bfr[4];
#pragma unroll
    for (int mt = 0; mt < 2; ++mt) {
      int rowb = (wm * 32 + mt * 16 + lq) * 40 + (half << 3);
#pragma unroll
      for (int v = 0; v < 8; ++v) {
        int kk = ((v & 4) << 2) | ((v & 3) << 1);   // A-frag K mapping
        af[mt].u[v] = *(const unsigned*)&As[buf][rowb + kk];
      }
    }
#pragma unroll
    for (int nt = 0; nt < 4; ++nt) {
      int colb = (half << 3) * 136 + wn * 64 + nt * 16 + lq;
#pragma unroll
      for (int v = 0; v < 8; ++v) {
        int kk = ((v & 4) << 2) | ((v & 3) << 1);
        bfr[nt].u[v] = (unsigned)Bs[buf][colb + kk * 136] |
                       ((unsigned)Bs[buf][colb + (kk + 1) * 136] << 16);
      }
    }
#pragma unroll
    for (int mt = 0; mt < 2; ++mt)
#pragma unroll
      for (int nt = 0; nt < 4; ++nt)
        c[mt][nt] = __builtin_amdgcn_wmma_f32_16x16x32_bf16(
            false, af[mt].v, false, bfr[nt].v, (short)0, c[mt][nt], false, false);
  };

  const int nk = K >> 5;
  issue(0, 0);
  for (int i = 0; i < nk - 1; ++i) {
    issue((i + 1) << 5, (i + 1) & 1);   // prefetch next tile
    wait_async4();                      // tile i resident (in-order completion)
    __syncthreads();
    compute(i & 1);
    __syncthreads();                    // all waves done reading buf i&1
  }
  wait_async0();
  __syncthreads();
  compute((nk - 1) & 1);

  // epilogue: bias add, optional row mask, store (coalesced across lanes)
#pragma unroll
  for (int mt = 0; mt < 2; ++mt) {
#pragma unroll
    for (int v = 0; v < 8; ++v) {
      int m = mBase + wm * 32 + mt * 16 + (half << 3) + v;
      float mk = 1.0f;
      if (MASKED) mk = rowmask[m] ? 1.0f : 0.0f;
#pragma unroll
      for (int nt = 0; nt < 4; ++nt) {
        int n = nBase + wn * 64 + nt * 16 + lq;
        float val = (c[mt][nt][v] + bias[n]) * mk;
        if (OUTBF16)
          ((bf16*)Out)[(size_t)m * ldo + n] = (bf16)val;
        else
          ((float*)Out)[(size_t)m * ldo + n] = val;
      }
    }
  }
}

// ------------------------------ attention ----------------------------------
// Two-pass softmax; block = 2 waves, each wave owns 16 queries and a 32 KB
// LDS slab holding its masked/scaled S^T row-block (16 x 512, fp32).
// S^T = K*Q^T so the per-block K A-fragment uses aligned dword loads and
// softmax stats live per-lane (lane = query): no cross-lane traffic in the
// hot loops. Key mask applied via wave32 __ballot (pure VALU test).

__global__ __launch_bounds__(64) void attn_kernel(
    const bf16* __restrict__ Qp, const bf16* __restrict__ Kp,
    const bf16* __restrict__ Vp, const int* __restrict__ kmask,
    bf16* __restrict__ Out, int colBase) {
  __shared__ __align__(16) float Sbuf[2][16 * 512];   // 64 KB
  const int bx = blockIdx.x;
  const int qc = bx & 15, h = (bx >> 4) & 15, b = bx >> 8;
  const int tid = threadIdx.x;
  const int lane = tid & 31, wave = tid >> 5;
  const int lq = lane & 15, half = lane >> 4;
  const int q0 = qc * 32 + wave * 16;
  const size_t rowQ0 = (size_t)(b * NO_ + q0);
  const int hc = h * DH_;
  const unsigned short* Qu = (const unsigned short*)Qp;
  float* myS = &Sbuf[wave][0];

  const v8f zero = {0.f, 0.f, 0.f, 0.f, 0.f, 0.f, 0.f, 0.f};

  // Q^T B-fragments: s=0 feats [0,32), s=1 feats [32,64)
  FragBF qf[2];
#pragma unroll
  for (int s = 0; s < 2; ++s) {
    int feat = s * 32 + (half << 3);
#pragma unroll
    for (int v = 0; v < 8; ++v) {
      int kk = ((v & 4) << 2) | ((v & 3) << 1);
      qf[s].u[v] = *(const unsigned*)&Qu[(rowQ0 + lq) * O_ + hc + feat + kk];
    }
  }

  // ---- pass A: S^T = K*Q^T, mask+scale, spill to LDS, track row max ----
  float mx = -__builtin_inff();
  for (int kb = 0; kb < NO_ / 32; ++kb) {
    const int k0 = kb * 32;
    unsigned mv = (unsigned)__ballot(kmask[b * NO_ + k0 + lane] != 0);
#pragma unroll
    for (int t = 0; t < 2; ++t) {
      FragBF kf0, kf1;
      const size_t krow =
          (size_t)(b * NO_ + k0 + t * 16 + lq) * O_ + hc + (half << 3);
#pragma unroll
      for (int v = 0; v < 8; ++v) {
        int kk = ((v & 4) << 2) | ((v & 3) << 1);
        kf0.u[v] = *(const unsigned*)(Kp + krow + kk);
        kf1.u[v] = *(const unsigned*)(Kp + krow + 32 + kk);
      }
      v8f s = __builtin_amdgcn_wmma_f32_16x16x32_bf16(
          false, kf0.v, false, qf[0].v, (short)0, zero, false, false);
      s = __builtin_amdgcn_wmma_f32_16x16x32_bf16(
          false, kf1.v, false, qf[1].v, (short)0, s, false, false);
      // masked_fill(key) then *SCALE (reference order), paired b64 spills
#pragma unroll
      for (int vp = 0; vp < 4; ++vp) {
        int lk = t * 16 + (half << 3) + 2 * vp;
        float s0 = (((mv >> lk) & 1u) ? s[2 * vp] : NEGV) * SCALE_;
        float s1 = (((mv >> (lk + 1)) & 1u) ? s[2 * vp + 1] : NEGV) * SCALE_;
        mx = fmaxf(mx, fmaxf(s0, s1));
        float2 pr; pr.x = s0; pr.y = s1;
        *(float2*)&myS[lq * 512 + k0 + lk] = pr;
      }
    }
  }
  mx = fmaxf(mx, __shfl_xor(mx, 16, 32));  // final max over 512 keys (per q)

  // ---- pass B: P = exp(S - m) (lane = query, stats need no shuffles), O += P*V
  v8f o[4];
#pragma unroll
  for (int i = 0; i < 4; ++i) o[i] = zero;
  float psum = 0.0f;
  for (int kb = 0; kb < NO_ / 32; ++kb) {
    const int k0 = kb * 32;
    FragBF pa;
#pragma unroll
    for (int v = 0; v < 8; ++v) {
      int kk = ((v & 4) << 2) | ((v & 3) << 1) | (half << 3);
      float2 sp = *(const float2*)&myS[lq * 512 + k0 + kk];
      float p0 = __expf(sp.x - mx);
      float p1 = __expf(sp.y - mx);
      psum += p0 + p1;
      pa.u[v] = (unsigned)f2bfu(p0) | ((unsigned)f2bfu(p1) << 16);
    }
#pragma unroll
    for (int ft = 0; ft < 4; ++ft) {
      FragBF vf;
      const size_t vrow = (size_t)(b * NO_ + k0 + (half << 3)) * O_ + hc + ft * 16 + lq;
      const unsigned short* Vu = (const unsigned short*)Vp;
#pragma unroll
      for (int v = 0; v < 8; ++v) {
        int kk = ((v & 4) << 2) | ((v & 3) << 1);
        vf.u[v] = (unsigned)Vu[vrow + (size_t)kk * O_] |
                  ((unsigned)Vu[vrow + (size_t)(kk + 1) * O_] << 16);
      }
      o[ft] = __builtin_amdgcn_wmma_f32_16x16x32_bf16(
          false, pa.v, false, vf.v, (short)0, o[ft], false, false);
    }
  }
  psum += __shfl_xor(psum, 16, 32);

  // normalize (broadcast l to row-form once) and store into cat
#pragma unroll
  for (int v = 0; v < 8; ++v) {
    int m = (half << 3) + v;
    float l = __shfl(psum, m, 16);
    float inv = 1.0f / l;
    size_t row = (size_t)(b * NO_ + q0 + m);
#pragma unroll
    for (int ft = 0; ft < 4; ++ft) {
      float val = o[ft][v] * inv;
      Out[row * 3072 + colBase + hc + ft * 16 + lq] = (bf16)val;
    }
  }
}

// ------------------------------ orchestration ------------------------------

extern "C" void kernel_launch(void* const* d_in, const int* in_sizes, int n_in,
                              void* d_out, int out_size, void* d_ws,
                              size_t ws_size, hipStream_t stream) {
  (void)in_sizes; (void)n_in; (void)out_size; (void)ws_size;
  const float* v      = (const float*)d_in[0];
  const float* q      = (const float*)d_in[1];
  const float* k      = (const float*)d_in[2];
  const int*   v_mask = (const int*)d_in[3];
  const int*   q_mask = (const int*)d_in[4];
  const int*   k_mask = (const int*)d_in[5];
  const float* Wv     = (const float*)d_in[6];
  const float* bv     = (const float*)d_in[7];
  const float* Wq     = (const float*)d_in[8];
  const float* bq     = (const float*)d_in[9];
  const float* Wk     = (const float*)d_in[10];
  const float* bk     = (const float*)d_in[11];
  const float* Wvo    = (const float*)d_in[12];
  const float* bvo    = (const float*)d_in[13];
  const float* Wko    = (const float*)d_in[14];
  const float* bko    = (const float*)d_in[15];

  char* ws = (char*)d_ws;
  size_t off = 0;
  auto alloc = [&](size_t elems) {
    bf16* p = (bf16*)(ws + off);
    off += ((elems * sizeof(bf16)) + 255) & ~(size_t)255;
    return p;
  };
  bf16* Wv_bf  = alloc((size_t)1024 * 3072);
  bf16* Wq_bf  = alloc((size_t)1024 * 3072);
  bf16* Wk_bf  = alloc((size_t)1024 * 3072);
  bf16* Wvo_bf = alloc((size_t)2048 * 1024);
  bf16* Wko_bf = alloc((size_t)3072 * 1024);
  bf16* a_v    = alloc((size_t)M_ * 1024);
  bf16* a_q    = alloc((size_t)M_ * 1024);
  bf16* a_k    = alloc((size_t)M_ * 1024);
  bf16* vq_buf = alloc((size_t)M_ * 1024);  // Q for attn1 (v_q)
  bf16* qk_buf = alloc((size_t)M_ * 1024);  // K for attn1 (q_k)
  bf16* qv_buf = alloc((size_t)M_ * 1024);  // V for attn1 (q_v)
  bf16* kk_buf = alloc((size_t)M_ * 1024);  // K for attn2 (k_k)
  bf16* kq_buf = alloc((size_t)M_ * 1024);  // Q and V for attn2 (k_q)
  bf16* cat    = alloc((size_t)M_ * 3072);  // [v | vq_update | vk_update]

  const int T = 256;
  cvt_bf16_kernel<<<(M_ * 1024) / 4 / T, T, 0, stream>>>(v, a_v, M_ * 1024, 1);
  cvt_bf16_kernel<<<(M_ * 1024) / 4 / T, T, 0, stream>>>(q, a_q, M_ * 1024, 1);
  cvt_bf16_kernel<<<(M_ * 1024) / 4 / T, T, 0, stream>>>(k, a_k, M_ * 1024, 1);
  cvt_bf16_kernel<<<(1024 * 3072) / 4 / T, T, 0, stream>>>(Wv, Wv_bf, 1024 * 3072, 0);
  cvt_bf16_kernel<<<(1024 * 3072) / 4 / T, T, 0, stream>>>(Wq, Wq_bf, 1024 * 3072, 0);
  cvt_bf16_kernel<<<(1024 * 3072) / 4 / T, T, 0, stream>>>(Wk, Wk_bf, 1024 * 3072, 0);
  cvt_bf16_kernel<<<(2048 * 1024) / 4 / T, T, 0, stream>>>(Wvo, Wvo_bf, 2048 * 1024, 0);
  cvt_bf16_kernel<<<(3072 * 1024) / 4 / T, T, 0, stream>>>(Wko, Wko_bf, 3072 * 1024, 0);
  cvt_v_cat_kernel<<<(M_ * 1024) / 4 / T, T, 0, stream>>>(v, cat);

  // 5 projection slices (only the thirds the reference actually consumes)
  dim3 ggrid(8, 32);
  gemm_bf16_kernel<true, true><<<ggrid, 256, 0, stream>>>(
      a_v, 1024, Wv_bf + 1024, 3072, bv + 1024, v_mask, vq_buf, 1024, 1024);
  gemm_bf16_kernel<true, true><<<ggrid, 256, 0, stream>>>(
      a_q, 1024, Wq_bf + 0,    3072, bq + 0,    q_mask, qk_buf, 1024, 1024);
  gemm_bf16_kernel<true, true><<<ggrid, 256, 0, stream>>>(
      a_q, 1024, Wq_bf + 2048, 3072, bq + 2048, q_mask, qv_buf, 1024, 1024);
  gemm_bf16_kernel<true, true><<<ggrid, 256, 0, stream>>>(
      a_k, 1024, Wk_bf + 0,    3072, bk + 0,    k_mask, kk_buf, 1024, 1024);
  gemm_bf16_kernel<true, true><<<ggrid, 256, 0, stream>>>(
      a_k, 1024, Wk_bf + 1024, 3072, bk + 1024, k_mask, kq_buf, 1024, 1024);

  // attentions -> cat[:,1024:2048] and cat[:,2048:3072]
  attn_kernel<<<2048, 64, 0, stream>>>(vq_buf, qk_buf, qv_buf, q_mask, cat, 1024);
  attn_kernel<<<2048, 64, 0, stream>>>(kq_buf, kk_buf, kq_buf, k_mask, cat, 2048);

  // output GEMMs: updated_V (Wko, K=3072) first, then updated_v (Wvo, K=2048)
  gemm_bf16_kernel<false, false><<<ggrid, 256, 0, stream>>>(
      cat, 3072, Wko_bf, 1024, bko, nullptr, d_out, 1024, 3072);
  gemm_bf16_kernel<false, false><<<ggrid, 256, 0, stream>>>(
      cat, 3072, Wvo_bf, 1024, bvo, nullptr,
      (float*)d_out + (size_t)M_ * 1024, 1024, 2048);
}
